// MyceliumEdgeConv_72078141162164
// MI455X (gfx1250) — matrix operand order repaired and checked by reading, not compiled, not run on hardware
//
#include <hip/hip_runtime.h>

typedef __attribute__((ext_vector_type(16))) __bf16 v16bf;
typedef __attribute__((ext_vector_type(8)))  float  v8f;
typedef __attribute__((ext_vector_type(4)))  unsigned int v4u;
typedef __attribute__((ext_vector_type(8)))  int    v8i_;
typedef __attribute__((ext_vector_type(4)))  int    v4i_;

#define IN_CH  128
#define OUT_CH 128
// K_total = 2*IN_CH = 256 ; K-steps of 32 -> 8 ; N-tiles of 16 -> 8

#if __has_builtin(__builtin_amdgcn_tensor_load_to_lds)
#define USE_TDM 1
#else
#define USE_TDM 0
#endif

union FragU { uint4 u[2]; v16bf v; };

// Build W1 (f32, [256][128] row-major) into the bf16 B-fragment-swizzled blob:
// blob[((ks*8 + nt)*32 + lane)*16 + j] = bf16( W1[(ks*32 + (j>>3)*16 + (lane>>4)*8 + (j&7))*128 + nt*16 + (lane&15)] )
__global__ void swizzleW1_kernel(const float* __restrict__ W1, __bf16* __restrict__ blob) {
    for (int g = threadIdx.x; g < 32768; g += blockDim.x) {
        int j    = g & 15;
        int lane = (g >> 4) & 31;
        int nt   = (g >> 9) & 7;
        int ks   = g >> 12;
        int N    = nt * 16 + (lane & 15);
        int K    = ks * 32 + ((j >> 3) << 4) + ((lane >> 4) << 3) + (j & 7);
        blob[g]  = (__bf16)W1[K * OUT_CH + N];
    }
}

struct bh4 { __bf16 v[4]; };

__global__ void cvtX_kernel(const float* __restrict__ x, __bf16* __restrict__ xb, int n) {
    int idx = (blockIdx.x * blockDim.x + threadIdx.x) * 4;
    if (idx < n) {
        float4 f = *(const float4*)(x + idx);
        bh4 o;
        o.v[0] = (__bf16)f.x; o.v[1] = (__bf16)f.y;
        o.v[2] = (__bf16)f.z; o.v[3] = (__bf16)f.w;
        *(bh4*)(xb + idx) = o;
    }
}

template <bool XBF>
__global__ __launch_bounds__(256)
void edgeconv_wmma_kernel(const float* __restrict__ xf,
                          const __bf16* __restrict__ xb,
                          const int* __restrict__ ei,
                          const float* __restrict__ b1,
                          const float* __restrict__ W2,
                          const float* __restrict__ b2,
                          const uint4* __restrict__ blob,
                          float* __restrict__ out,
                          int E) {
    __shared__ uint4 sW[4096];  // 64 KB: pre-swizzled bf16 W1 B-fragments

    const int tid  = threadIdx.x;
    const int lane = tid & 31;
    const int wave = tid >> 5;

#if USE_TDM
    // One TDM descriptor per block DMAs the 64 KB W1 blob L2 -> LDS.
    if (wave == 0) {
        const unsigned long long ga = (unsigned long long)(uintptr_t)blob;
        const unsigned int ldsAddr  = (unsigned int)(uintptr_t)&sW[0]; // low 32b = LDS offset
        v4u g0;
        g0[0] = 1u;                                           // count=1 (valid user D#)
        g0[1] = ldsAddr;                                      // lds_addr [63:32]
        g0[2] = (unsigned int)(ga & 0xffffffffu);             // global_addr [95:64]
        g0[3] = (unsigned int)((ga >> 32) & 0x01ffffffu)      // global_addr [120:96]
              | (2u << 30);                                   // type=2 ("image") [127:126]
        v8i_ g1;
        g1[0] = 0x00020000;          // workgroup_mask=0, data_size=2 (4-byte elements)
        g1[1] = (int)(16384u << 16); // tensor_dim0[15:0]=16384 dwords (<<16); barrier_addr=0
        g1[2] = (int)((0u) | (1u << 16));       // tensor_dim0[31:16]=0 ; tensor_dim1[15:0]=1
        g1[3] = (int)(16384u << 16);            // tensor_dim1[31:16]=0 ; tile_dim0=16384
        g1[4] = 1;                              // tile_dim1=1 ; tile_dim2=0
        g1[5] = 16384;                          // tensor_dim0_stride[31:0]
        g1[6] = (int)(16384u << 16);            // stride0[47:32]=0 ; stride1[15:0]=16384
        g1[7] = 0;                              // stride1[47:16]=0
        v4i_ z4 = {0, 0, 0, 0};
        v8i_ z8 = {0, 0, 0, 0, 0, 0, 0, 0};
        // 6-arg form (this toolchain): (g0, g1, g2, g3, g4, cpol)
        __builtin_amdgcn_tensor_load_to_lds(g0, g1, z4, z4, z8, 0);
        __builtin_amdgcn_s_wait_tensorcnt(0);   // TDM done before barrier publishes LDS
    }
    __syncthreads();
#else
#pragma unroll
    for (int k = 0; k < 16; ++k)
        sW[tid + k * 256] = blob[tid + k * 256];
    __syncthreads();
#endif

    const int e0 = blockIdx.x * 128 + wave * 16;
    if (e0 >= E) return;                 // uniform per wave: EXEC stays all-1s
    const int m      = lane & 15;        // edge row within tile / out-channel lane
    const int half16 = lane >> 4;        // which 16-lane half

    int edge = e0 + m;
    if (edge >= E) edge = E - 1;         // clamp (keeps all lanes live for WMMA)
    const int nrow = ei[edge];
    const int ncol = ei[E + edge];

    v8f acc[8] = {};                     // 8 N-tiles x (16x16 f32 / 32 lanes)

#pragma unroll
    for (int ks = 0; ks < 8; ++ks) {
        const int node = (ks < 4) ? nrow : ncol;
        const int feat = (ks & 3) * 32 + half16 * 8;  // A-layout: lane<16 -> K 0..7/16..23 ; lane>=16 -> K 8..15/24..31
        FragU a;
        if (XBF) {
            const __bf16* p = xb + (size_t)node * IN_CH + feat;
            a.u[0] = *(const uint4*)(p);
            a.u[1] = *(const uint4*)(p + 16);
        } else {
            const float* p = xf + (size_t)node * IN_CH + feat;
#pragma unroll
            for (int j = 0; j < 8; ++j) a.v[j]     = (__bf16)p[j];
#pragma unroll
            for (int j = 0; j < 8; ++j) a.v[8 + j] = (__bf16)p[16 + j];
        }
#pragma unroll
        for (int nt = 0; nt < 8; ++nt) {
            FragU b;
            const int bi = ((ks * 8 + nt) * 32 + lane) * 2;
            b.u[0] = sW[bi];
            b.u[1] = sW[bi + 1];
            acc[nt] = __builtin_amdgcn_wmma_f32_16x16x32_bf16(
                false, a.v, false, b.v, (short)0, acc[nt], false, false);
        }
    }

    // Fused layer 2: relu(h + b1) . W2, reduced across the 16 out-channel lanes.
    float partial[8];
#pragma unroll
    for (int r = 0; r < 8; ++r) partial[r] = 0.0f;

#pragma unroll
    for (int nt = 0; nt < 8; ++nt) {
        const int ch = nt * 16 + m;
        const float bb = b1[ch];
        const float w  = W2[ch];
#pragma unroll
        for (int r = 0; r < 8; ++r) {
            float h = acc[nt][r] + bb;
            h = h > 0.0f ? h : 0.0f;
            partial[r] += h * w;
        }
    }

#pragma unroll
    for (int r = 0; r < 8; ++r) {
#pragma unroll
        for (int mask = 1; mask <= 8; mask <<= 1)
            partial[r] += __shfl_xor(partial[r], mask, 32);
    }

    const float bias2 = b2[0];
    if (m == 0) {                         // lanes 0 and 16 write 8 edges each
#pragma unroll
        for (int r = 0; r < 8; ++r) {
            const int em = e0 + r + half16 * 8;
            if (em < E) {
                const float t = partial[r] + bias2;
                out[em] = 1.0f / (1.0f + __expf(-t));
            }
        }
    }
}

extern "C" void kernel_launch(void* const* d_in, const int* in_sizes, int n_in,
                              void* d_out, int out_size, void* d_ws, size_t ws_size,
                              hipStream_t stream) {
    const float* x  = (const float*)d_in[0];
    const int*   ei = (const int*)d_in[1];
    const float* W1 = (const float*)d_in[2];
    const float* b1 = (const float*)d_in[3];
    const float* W2 = (const float*)d_in[4];
    const float* b2 = (const float*)d_in[5];
    float* out = (float*)d_out;

    const int nodes = in_sizes[0] / IN_CH;
    const int E     = in_sizes[1] / 2;

    const size_t blobBytes = 32768 * sizeof(__bf16);   // 64 KB swizzled W1
    __bf16* blob = (__bf16*)d_ws;
    __bf16* xb   = (__bf16*)((char*)d_ws + blobBytes);
    const size_t xbBytes = (size_t)nodes * IN_CH * sizeof(__bf16);
    const bool useXb = ws_size >= blobBytes + xbBytes;

    swizzleW1_kernel<<<1, 256, 0, stream>>>(W1, blob);

    const int grid = (E + 127) / 128;
    if (useXb) {
        const int n = nodes * IN_CH;
        cvtX_kernel<<<(n / 4 + 255) / 256, 256, 0, stream>>>(x, xb, n);
        edgeconv_wmma_kernel<true><<<grid, 256, 0, stream>>>(
            x, xb, ei, b1, W2, b2, (const uint4*)blob, out, E);
    } else {
        edgeconv_wmma_kernel<false><<<grid, 256, 0, stream>>>(
            x, xb, ei, b1, W2, b2, (const uint4*)blob, out, E);
    }
}
